// CausalAttention_7722351199028
// MI455X (gfx1250) — matrix operand order repaired and checked by reading, not compiled, not run on hardware
//
#include <hip/hip_runtime.h>
#include <hip/hip_bf16.h>

// ---------------------------------------------------------------------------
// Causal attention for MI455X (gfx1250, wave32, WMMA bf16 16x16x32).
//
// Pipeline (intermediates in d_ws; 144 MB total -> resident in 192 MB L2):
//   [K1] Q/K/V = x @ W{q,k,v}        (f32 in -> bf16 out)       v_wmma bf16
//   [K2] S     = scale * Q @ K^T     (lower-triangular tiles)   v_wmma bf16
//   [K3] P     = causal row-softmax(S)  (f32 math, bf16 out)    VALU
//   [K4] O     = P @ V               (causal-trimmed K loop)    v_wmma bf16
//
// All GEMMs: 128x128 tile / 256 threads (8 waves), 32-deep K chunks,
// double-buffered LDS with register staging so global loads for chunk i+1
// overlap the 8 WMMAs of chunk i. One s_barrier per chunk.
// ---------------------------------------------------------------------------

typedef __attribute__((ext_vector_type(16))) __bf16 v16bf;
typedef __attribute__((ext_vector_type(8)))  __bf16 v8bf;
typedef __attribute__((ext_vector_type(8)))  float  v8f;
typedef __attribute__((ext_vector_type(4)))  float  v4f;

#define B_  4
#define N_  2048
#define D_  1024

#define BM  128      // output tile rows per workgroup
#define BN  128      // output tile cols per workgroup
#define BK  32       // WMMA K depth per chunk
#define LDT 40       // padded LDS row stride in bf16 (80 B, 16B-aligned)

// A-fragment (16x32 bf16): lane m=L&15; halves 0..7 -> K=base..base+7,
// halves 8..15 -> K=base+16..base+23, base = (L<16)?0:8.   (ISA 7.12.2)
static __device__ __forceinline__ v16bf ld_frag_a(const __bf16* row, int lane) {
  const int base = (lane & 16) ? 8 : 0;
  v8bf lo = *(const v8bf*)(row + base);
  v8bf hi = *(const v8bf*)(row + base + 16);
  v16bf r;
#pragma unroll
  for (int i = 0; i < 8; ++i) { r[i] = lo[i]; r[i + 8] = hi[i]; }
  return r;
}

// B-fragment (32x16 bf16), LDS tile stored N-major (row = n, 32 K's per row):
// lane n=L&15; halves 0..7 -> K=base..base+7, halves 8..15 -> K=base+16..base+23, base = (L<16)?0:8.
static __device__ __forceinline__ v16bf ld_frag_b(const __bf16* row, int lane) {
  const int base = (lane & 16) ? 8 : 0;
  v8bf lo = *(const v8bf*)(row + base);
  v8bf hi = *(const v8bf*)(row + base + 16);
  v16bf r;
#pragma unroll
  for (int i = 0; i < 8; ++i) { r[i] = lo[i]; r[i + 8] = hi[i]; }
  return r;
}

static __device__ __forceinline__ v8f wmma_bf16(v16bf a, v16bf b, v8f c) {
  return __builtin_amdgcn_wmma_f32_16x16x32_bf16(false, a, false, b, (short)0, c,
                                                 false, false);
}

// One K-chunk of the 2x4-tile wave GEMM: 6 ds_load_b128 pairs -> 8 WMMAs.
static __device__ __forceinline__ void mma_chunk(const __bf16* A, const __bf16* Bt,
                                                 int lane, int wm, int wn,
                                                 v8f acc[2][4]) {
  v16bf af[2], bfv[4];
#pragma unroll
  for (int tm = 0; tm < 2; ++tm)
    af[tm] = ld_frag_a(&A[(wm + tm * 16 + (lane & 15)) * LDT], lane);
#pragma unroll
  for (int tn = 0; tn < 4; ++tn)
    bfv[tn] = ld_frag_b(&Bt[(wn + tn * 16 + (lane & 15)) * LDT], lane);
#pragma unroll
  for (int tm = 0; tm < 2; ++tm)
#pragma unroll
    for (int tn = 0; tn < 4; ++tn)
      acc[tm][tn] = wmma_bf16(af[tm], bfv[tn], acc[tm][tn]);
}

// ---------------------------------------------------------------------------
// K1: Q/K/V projection GEMM. grid = (M/BM, D/BN, 3), block = 256.
// ---------------------------------------------------------------------------
__global__ __launch_bounds__(256) void qkv_kernel(
    const float* __restrict__ x,
    const float* __restrict__ Wq, const float* __restrict__ Wk,
    const float* __restrict__ Wv,
    __bf16* __restrict__ Qb, __bf16* __restrict__ Kb, __bf16* __restrict__ Vb) {
  const float* W;
  __bf16* out;
  if (blockIdx.z == 0)      { W = Wq; out = Qb; }
  else if (blockIdx.z == 1) { W = Wk; out = Kb; }
  else                      { W = Wv; out = Vb; }

  const int m0 = blockIdx.x * BM, n0 = blockIdx.y * BN;
  __shared__ __bf16 sA[2][BM * LDT];   // x tile, row-major [m][k]
  __shared__ __bf16 sB[2][BN * LDT];   // W tile, transposed to [n][k]
  const int tid = threadIdx.x, lane = tid & 31, wid = tid >> 5;
  const int wm = (wid & 3) * 32, wn = (wid >> 2) * 64;

  v4f ax[2][2];  // staged x:  2 units x 8 floats
  v4f wx[2][2];  // staged W:  2 units x 8 floats

  auto g_load = [&](int k0) {
#pragma unroll
    for (int j = 0; j < 2; ++j) {
      const int u = tid + j * 256;
      {
        const int r = u >> 2, c8 = (u & 3) * 8;
        const float* p = &x[(size_t)(m0 + r) * D_ + k0 + c8];
        ax[j][0] = *(const v4f*)p;
        ax[j][1] = *(const v4f*)(p + 4);
        if (k0 + 2 * BK < D_) __builtin_prefetch(p + 2 * BK, 0, 1);
      }
      {
        const int kk = u >> 4, nn8 = (u & 15) * 8;
        const float* p = &W[(size_t)(k0 + kk) * D_ + n0 + nn8];
        wx[j][0] = *(const v4f*)p;
        wx[j][1] = *(const v4f*)(p + 4);
      }
    }
  };
  auto s_store = [&](int buf) {
#pragma unroll
    for (int j = 0; j < 2; ++j) {
      const int u = tid + j * 256;
      {
        const int r = u >> 2, c8 = (u & 3) * 8;
        v8bf t;
#pragma unroll
        for (int i = 0; i < 4; ++i) {
          t[i]     = (__bf16)ax[j][0][i];
          t[i + 4] = (__bf16)ax[j][1][i];
        }
        *(v8bf*)&sA[buf][r * LDT + c8] = t;
      }
      {
        const int kk = u >> 4, nn8 = (u & 15) * 8;   // transpose scatter
#pragma unroll
        for (int i = 0; i < 8; ++i)
          sB[buf][(nn8 + i) * LDT + kk] =
              (__bf16)(i < 4 ? wx[j][0][i] : wx[j][1][i - 4]);
      }
    }
  };

  v8f acc[2][4];
  const v8f z = {0.f, 0.f, 0.f, 0.f, 0.f, 0.f, 0.f, 0.f};
#pragma unroll
  for (int i = 0; i < 2; ++i)
#pragma unroll
    for (int j = 0; j < 4; ++j) acc[i][j] = z;

  g_load(0);
  s_store(0);
  __syncthreads();
  const int nch = D_ / BK;
  for (int ch = 0; ch < nch; ++ch) {
    const int cur = ch & 1;
    if (ch + 1 < nch) g_load((ch + 1) * BK);          // overlap with WMMAs
    mma_chunk(sA[cur], sB[cur], lane, wm, wn, acc);
    if (ch + 1 < nch) s_store(cur ^ 1);
    __syncthreads();
  }

  const int mo = (lane & 16) ? 8 : 0, nc = lane & 15;
#pragma unroll
  for (int tm = 0; tm < 2; ++tm)
#pragma unroll
    for (int tn = 0; tn < 4; ++tn)
#pragma unroll
      for (int v = 0; v < 8; ++v) {
        const int m = m0 + wm + tm * 16 + v + mo;
        const int n = n0 + wn + tn * 16 + nc;
        out[(size_t)m * D_ + n] = (__bf16)acc[tm][tn][v];
      }
}

// ---------------------------------------------------------------------------
// K2: S = scale * Q @ K^T (per batch), lower-triangular tiles only.
// grid = (N/BM, N/BN, B), block = 256.  K row-major *is* the N-major B tile.
// ---------------------------------------------------------------------------
__global__ __launch_bounds__(256) void scores_kernel(
    const __bf16* __restrict__ Qb, const __bf16* __restrict__ Kb,
    float* __restrict__ S) {
  const int m0 = blockIdx.x * BM, n0 = blockIdx.y * BN;
  if (n0 > m0 + (BM - 1)) return;  // fully masked tile: softmax never reads it
  const int b = blockIdx.z;
  const __bf16* Q  = Qb + (size_t)b * N_ * D_;
  const __bf16* Kp = Kb + (size_t)b * N_ * D_;
  float* Sb = S + (size_t)b * N_ * N_;

  __shared__ __bf16 sA[2][BM * LDT];
  __shared__ __bf16 sB[2][BN * LDT];
  const int tid = threadIdx.x, lane = tid & 31, wid = tid >> 5;
  const int wm = (wid & 3) * 32, wn = (wid >> 2) * 64;

  v8bf qa[2], ka[2];  // staged tiles (8 bf16 per unit)

  auto g_load = [&](int k0) {
#pragma unroll
    for (int j = 0; j < 2; ++j) {
      const int u = tid + j * 256;
      const int r = u >> 2, c8 = (u & 3) * 8;
      const __bf16* pq = &Q[(size_t)(m0 + r) * D_ + k0 + c8];
      qa[j] = *(const v8bf*)pq;
      ka[j] = *(const v8bf*)&Kp[(size_t)(n0 + r) * D_ + k0 + c8];
      if (k0 + 2 * BK < D_) __builtin_prefetch(pq + 2 * BK, 0, 1);
    }
  };
  auto s_store = [&](int buf) {
#pragma unroll
    for (int j = 0; j < 2; ++j) {
      const int u = tid + j * 256;
      const int r = u >> 2, c8 = (u & 3) * 8;
      *(v8bf*)&sA[buf][r * LDT + c8] = qa[j];
      *(v8bf*)&sB[buf][r * LDT + c8] = ka[j];
    }
  };

  v8f acc[2][4];
  const v8f z = {0.f, 0.f, 0.f, 0.f, 0.f, 0.f, 0.f, 0.f};
#pragma unroll
  for (int i = 0; i < 2; ++i)
#pragma unroll
    for (int j = 0; j < 4; ++j) acc[i][j] = z;

  g_load(0);
  s_store(0);
  __syncthreads();
  const int nch = D_ / BK;
  for (int ch = 0; ch < nch; ++ch) {
    const int cur = ch & 1;
    if (ch + 1 < nch) g_load((ch + 1) * BK);
    mma_chunk(sA[cur], sB[cur], lane, wm, wn, acc);
    if (ch + 1 < nch) s_store(cur ^ 1);
    __syncthreads();
  }

  const float scale = 0.03125f;  // 1/sqrt(1024)
  const int mo = (lane & 16) ? 8 : 0, nc = lane & 15;
#pragma unroll
  for (int tm = 0; tm < 2; ++tm)
#pragma unroll
    for (int tn = 0; tn < 4; ++tn)
#pragma unroll
      for (int v = 0; v < 8; ++v) {
        const int m = m0 + wm + tm * 16 + v + mo;
        const int n = n0 + wn + tn * 16 + nc;
        // entries with n > m are garbage but never read (softmax reads 0..m)
        Sb[(size_t)m * N_ + n] = acc[tm][tn][v] * scale;
      }
}

// ---------------------------------------------------------------------------
// K3: causal row softmax. grid = (N, B), block = 256. One row per workgroup.
// Writes full 2048-wide bf16 row with explicit zeros past the diagonal so the
// P@V GEMM can run rectangular K-chunks.
// ---------------------------------------------------------------------------
__global__ __launch_bounds__(256) void softmax_kernel(
    const float* __restrict__ S, __bf16* __restrict__ P) {
  const int r = blockIdx.x, b = blockIdx.y, tid = threadIdx.x;
  const float* srow = S + ((size_t)b * N_ + r) * N_;
  __bf16* prow = P + ((size_t)b * N_ + r) * N_;
  const int len = r + 1;

  __shared__ float red[256];
  float mx = -1e30f;
  for (int n = tid; n < len; n += 256) mx = fmaxf(mx, srow[n]);
  red[tid] = mx;
  __syncthreads();
  for (int s = 128; s > 0; s >>= 1) {
    if (tid < s) red[tid] = fmaxf(red[tid], red[tid + s]);
    __syncthreads();
  }
  mx = red[0];
  __syncthreads();

  float sum = 0.f;
  for (int n = tid; n < len; n += 256) sum += __expf(srow[n] - mx);
  red[tid] = sum;
  __syncthreads();
  for (int s = 128; s > 0; s >>= 1) {
    if (tid < s) red[tid] += red[tid + s];
    __syncthreads();
  }
  const float inv = 1.0f / red[0];

  for (int n = tid; n < N_; n += 256) {
    const float p = (n < len) ? __expf(srow[n] - mx) * inv : 0.0f;
    prow[n] = (__bf16)p;
  }
}

// ---------------------------------------------------------------------------
// K4: O = P @ V (per batch), K loop trimmed at the causal diagonal.
// grid = (N/BM, D/BN, B), block = 256.
// ---------------------------------------------------------------------------
__global__ __launch_bounds__(256) void pv_kernel(
    const __bf16* __restrict__ P, const __bf16* __restrict__ Vb,
    float* __restrict__ O) {
  const int b = blockIdx.z;
  const int m0 = blockIdx.x * BM, n0 = blockIdx.y * BN;
  const __bf16* Pb = P + (size_t)b * N_ * N_;
  const __bf16* Vp = Vb + (size_t)b * N_ * D_;
  float* Ob = O + (size_t)b * N_ * D_;

  __shared__ __bf16 sA[2][BM * LDT];
  __shared__ __bf16 sB[2][BN * LDT];
  const int tid = threadIdx.x, lane = tid & 31, wid = tid >> 5;
  const int wm = (wid & 3) * 32, wn = (wid >> 2) * 64;

  const int kend = m0 + BM;       // P[m][k]==0 for k>m; tile bound m0+127
  const int nch = kend / BK;      // causal-trimmed chunk count

  v8bf pa[2], va[2];

  auto g_load = [&](int k0) {
#pragma unroll
    for (int j = 0; j < 2; ++j) {
      const int u = tid + j * 256;
      {
        const int r = u >> 2, c8 = (u & 3) * 8;
        const __bf16* pp = &Pb[(size_t)(m0 + r) * N_ + k0 + c8];
        pa[j] = *(const v8bf*)pp;
        if (k0 + 2 * BK < kend) __builtin_prefetch(pp + 2 * BK, 0, 1);
      }
      {
        const int kk = u >> 4, nn8 = (u & 15) * 8;
        va[j] = *(const v8bf*)&Vp[(size_t)(k0 + kk) * D_ + n0 + nn8];
      }
    }
  };
  auto s_store = [&](int buf) {
#pragma unroll
    for (int j = 0; j < 2; ++j) {
      const int u = tid + j * 256;
      {
        const int r = u >> 2, c8 = (u & 3) * 8;
        *(v8bf*)&sA[buf][r * LDT + c8] = pa[j];
      }
      {
        const int kk = u >> 4, nn8 = (u & 15) * 8;   // transpose scatter
#pragma unroll
        for (int i = 0; i < 8; ++i) sB[buf][(nn8 + i) * LDT + kk] = va[j][i];
      }
    }
  };

  v8f acc[2][4];
  const v8f z = {0.f, 0.f, 0.f, 0.f, 0.f, 0.f, 0.f, 0.f};
#pragma unroll
  for (int i = 0; i < 2; ++i)
#pragma unroll
    for (int j = 0; j < 4; ++j) acc[i][j] = z;

  g_load(0);
  s_store(0);
  __syncthreads();
  for (int ch = 0; ch < nch; ++ch) {
    const int cur = ch & 1;
    if (ch + 1 < nch) g_load((ch + 1) * BK);
    mma_chunk(sA[cur], sB[cur], lane, wm, wn, acc);
    if (ch + 1 < nch) s_store(cur ^ 1);
    __syncthreads();
  }

  const int mo = (lane & 16) ? 8 : 0, nc = lane & 15;
#pragma unroll
  for (int tm = 0; tm < 2; ++tm)
#pragma unroll
    for (int tn = 0; tn < 4; ++tn)
#pragma unroll
      for (int v = 0; v < 8; ++v) {
        const int m = m0 + wm + tm * 16 + v + mo;
        const int n = n0 + wn + tn * 16 + nc;
        Ob[(size_t)m * D_ + n] = acc[tm][tn][v];
      }
}

// ---------------------------------------------------------------------------
extern "C" void kernel_launch(void* const* d_in, const int* in_sizes, int n_in,
                              void* d_out, int out_size, void* d_ws,
                              size_t ws_size, hipStream_t stream) {
  (void)in_sizes; (void)n_in; (void)out_size; (void)ws_size;
  const float* x  = (const float*)d_in[0];
  const float* Wq = (const float*)d_in[1];
  const float* Wk = (const float*)d_in[2];
  const float* Wv = (const float*)d_in[3];
  float* O = (float*)d_out;

  // Workspace layout (bytes):
  //   Qb bf16 : [0,            16,777,216)
  //   Kb bf16 : [16,777,216,   33,554,432)
  //   Vb bf16 : [33,554,432,   50,331,648)
  //   S  f32  : [50,331,648,  117,440,512)
  //   P  bf16 : [117,440,512, 150,994,944)   -> 144 MB total, fits 192 MB L2
  char* ws = (char*)d_ws;
  const size_t qkv_elems = (size_t)B_ * N_ * D_;  // 8,388,608
  __bf16* Qb = (__bf16*)(ws);
  __bf16* Kb = (__bf16*)(ws + qkv_elems * 2);
  __bf16* Vb = (__bf16*)(ws + qkv_elems * 4);
  float*  S  = (float*)(ws + qkv_elems * 6);
  __bf16* P  = (__bf16*)(ws + qkv_elems * 6 + (size_t)B_ * N_ * N_ * 4);

  dim3 blk(256);
  qkv_kernel<<<dim3((B_ * N_) / BM, D_ / BN, 3), blk, 0, stream>>>(
      x, Wq, Wk, Wv, Qb, Kb, Vb);
  scores_kernel<<<dim3(N_ / BM, N_ / BN, B_), blk, 0, stream>>>(Qb, Kb, S);
  softmax_kernel<<<dim3(N_, B_), blk, 0, stream>>>(S, P);
  pv_kernel<<<dim3(N_ / BM, D_ / BN, B_), blk, 0, stream>>>(P, Vb, O);
}
